// G2ANet_1468878815419
// MI455X (gfx1250) — compile-verified, matrix operand order checked by reading, not run on hardware
//
#include <hip/hip_runtime.h>
#include <hip/hip_bf16.h>

typedef __attribute__((ext_vector_type(16))) _Float16 v16h;
typedef __attribute__((ext_vector_type(8)))  _Float16 v8h;
typedef __attribute__((ext_vector_type(4)))  _Float16 v4h;
typedef __attribute__((ext_vector_type(8)))  float    v8f;
typedef __attribute__((ext_vector_type(4)))  float    v4f;

#define NAG   32          // N_AGENTS
#define HID   64
#define ATT   32
#define OBSD  256
#define NA    20
#define NM    31          // N_AGENTS - 1
#define ROWS  16384       // B * N_AGENTS  (B = 512)
#define TAUF  0.01f

// ---------------------------------------------------------------------------
// CDNA5 wave32 WMMA 16x16x32 f16->f32 fragment helpers.
// A frag: lane 0-15 row M=lane, K={khi*8+0..7, 16+khi*8+0..7}, khi=lane>>4
//         -> two CONTIGUOUS 8-half runs per lane => 2x ds_load_b128.
// B frag: pre-swizzled weight tiles, lane's 16 halves contiguous
//         -> 2x ds_load_b128.
// C frag: VGPR e -> (M = e + 8*(lane>>4), N = lane&15).
// ---------------------------------------------------------------------------
__device__ __forceinline__ v16h load_a_row(const _Float16* rowp) {
  int khi = (threadIdx.x >> 4) & 1;
  const _Float16* p = rowp + khi * 8;
  union { v16h v; v8h h[2]; } u;
  u.h[0] = *(const v8h*)(p);
  u.h[1] = *(const v8h*)(p + 16);
  return u.v;
}

__device__ __forceinline__ v16h load_a_lds(const _Float16* A, int lda) {
  return load_a_row(A + (threadIdx.x & 15) * lda);
}

__device__ __forceinline__ v16h load_a_g32(const float* A, int lda) {
  int lane = threadIdx.x & 31;
  const float* p = A + (lane & 15) * lda + (lane >> 4) * 8;
  v16h a;
#pragma unroll
  for (int h = 0; h < 8; ++h) {
    a[h] = (_Float16)p[h];
    a[h + 8] = (_Float16)p[16 + h];
  }
  return a;
}

__device__ __forceinline__ v16h load_bfrag(const _Float16* tile) {
  const _Float16* p = tile + (threadIdx.x & 31) * 16;
  union { v16h v; v8h h[2]; } u;
  u.h[0] = *(const v8h*)(p);
  u.h[1] = *(const v8h*)(p + 8);
  return u.v;
}

// Stage a row-major KDIMxNDIM f32 weight matrix into fragment-tile layout:
// tile (kc,nt) occupies 512 halves; element (k,n) -> tile[lane*16 + h].
template <int KDIM, int NDIM>
__device__ void stage_w_frag(const float* __restrict__ W, _Float16* dst) {
  constexpr int nNT = NDIM / 16;
  for (int i = threadIdx.x; i < KDIM * NDIM; i += 256) {
    int k = i / NDIM, n = i % NDIM;
    int kc = k >> 5, kk = k & 31;
    int khi = (kk >> 3) & 1;
    int base = kk - khi * 8;                    // {0..7} u {16..23}
    int h = (base >= 16) ? base - 8 : base;
    int lane = (n & 15) + khi * 16;
    dst[(((kc * nNT) + (n >> 4)) << 9) + lane * 16 + h] = (_Float16)W[i];
  }
}

__device__ __forceinline__ v8f wmma16(v16h a, v16h b, v8f c) {
  return __builtin_amdgcn_wmma_f32_16x16x32_f16(false, a, false, b,
                                                (short)0, c, false, false);
}

__device__ __forceinline__ float sigm(float x) { return 1.f / (1.f + __expf(-x)); }

// ---------------------------------------------------------------------------
// Kernel 1: enc = relu(obs@W_enc+b) ; h_out = GRU(enc, h_prev) ; q,k,v proj
// 256 threads (8 waves), 128 rows per block, 128 blocks.
// Also emits an f16 copy of h_out so kernel 2 stages without conversions.
// ---------------------------------------------------------------------------
__global__ void k1_encode_gru_qkv(
    const float* __restrict__ obs, const float* __restrict__ hprev,
    const float* __restrict__ W_enc, const float* __restrict__ b_enc,
    const float* __restrict__ Wih_c, const float* __restrict__ Whh_c,
    const float* __restrict__ bih_c, const float* __restrict__ bhh_c,
    const float* __restrict__ Wq, const float* __restrict__ Wk,
    const float* __restrict__ Wv, const float* __restrict__ b_v,
    float* __restrict__ hout, _Float16* __restrict__ hout16,
    float* __restrict__ qo, float* __restrict__ ko, float* __restrict__ vo,
    float* __restrict__ dout_h) {
  extern __shared__ char smem[];
  _Float16* sWenc = (_Float16*)smem;          // 256*64  frag tiles (8x4)
  _Float16* sWih  = sWenc + OBSD * HID;       // 64*192  frag tiles (2x12)
  _Float16* sWhh  = sWih + HID * 192;         // 64*192
  _Float16* sWq   = sWhh + HID * 192;         // 64*32   frag tiles (2x2)
  _Float16* sWk   = sWq + HID * ATT;
  _Float16* sWv   = sWk + HID * ATT;
  _Float16* sEnc  = sWv + HID * ATT;          // 128*64 row-major
  _Float16* sH    = sEnc + 128 * HID;         // 128*64 row-major
  _Float16* sHout = sH + 128 * HID;           // 128*64 row-major
  float* sBe  = (float*)(sHout + 128 * HID);  // 64
  float* sBi  = sBe + HID;                    // 192
  float* sBh  = sBi + 192;                    // 192
  float* sBv  = sBh + 192;                    // 32

  const int tid = threadIdx.x;
  const int rowBase = blockIdx.x * 128;

  stage_w_frag<OBSD, HID>(W_enc, sWenc);
  stage_w_frag<HID, 192>(Wih_c, sWih);
  stage_w_frag<HID, 192>(Whh_c, sWhh);
  stage_w_frag<HID, ATT>(Wq, sWq);
  stage_w_frag<HID, ATT>(Wk, sWk);
  stage_w_frag<HID, ATT>(Wv, sWv);
  for (int i = tid; i < HID; i += 256) sBe[i] = b_enc[i];
  for (int i = tid; i < 192; i += 256) { sBi[i] = bih_c[i]; sBh[i] = bhh_c[i]; }
  for (int i = tid; i < ATT; i += 256) sBv[i] = b_v[i];
  for (int i4 = tid * 4; i4 < 128 * HID; i4 += 256 * 4) {
    v4f s = *(const v4f*)(hprev + (size_t)rowBase * HID + i4);
    v4h d; d[0]=(_Float16)s[0]; d[1]=(_Float16)s[1]; d[2]=(_Float16)s[2]; d[3]=(_Float16)s[3];
    *(v4h*)(sH + i4) = d;
  }
  __syncthreads();

  const int wave = tid >> 5, lane = tid & 31;
  const int rowLoc = wave * 16;
  const int gRow = rowBase + rowLoc;
  const int khi = lane >> 4, col = lane & 15;

  // ---- encoder: (16x256) @ (256x64), relu, -> sEnc (f16) ----
  for (int nt = 0; nt < 4; ++nt) {
    v8f acc = {};
#pragma unroll
    for (int kc = 0; kc < 8; ++kc) {
      v16h a = load_a_g32(obs + (size_t)gRow * OBSD + kc * 32, OBSD);
      acc = wmma16(a, load_bfrag(sWenc + ((kc * 4 + nt) << 9)), acc);
    }
#pragma unroll
    for (int e = 0; e < 8; ++e) {
      int r = rowLoc + e + khi * 8;
      float vvl = acc[e] + sBe[nt * 16 + col];
      sEnc[r * HID + nt * 16 + col] = (_Float16)(vvl > 0.f ? vvl : 0.f);
    }
  }

  // ---- comm GRU: gi = enc@Wih_c, gh = h@Whh_c, gate combine ----
  {
    v16h ae0 = load_a_lds(sEnc + rowLoc * HID, HID);
    v16h ae1 = load_a_lds(sEnc + rowLoc * HID + 32, HID);
    v16h ah0 = load_a_lds(sH + rowLoc * HID, HID);
    v16h ah1 = load_a_lds(sH + rowLoc * HID + 32, HID);
    for (int t = 0; t < 4; ++t) {
      v8f gR = {}, gZ = {}, gN = {}, hR = {}, hZ = {}, hN = {};
      gR = wmma16(ae0, load_bfrag(sWih + ((t) << 9)), gR);
      gZ = wmma16(ae0, load_bfrag(sWih + ((t + 4) << 9)), gZ);
      gN = wmma16(ae0, load_bfrag(sWih + ((t + 8) << 9)), gN);
      gR = wmma16(ae1, load_bfrag(sWih + ((12 + t) << 9)), gR);
      gZ = wmma16(ae1, load_bfrag(sWih + ((12 + t + 4) << 9)), gZ);
      gN = wmma16(ae1, load_bfrag(sWih + ((12 + t + 8) << 9)), gN);
      hR = wmma16(ah0, load_bfrag(sWhh + ((t) << 9)), hR);
      hZ = wmma16(ah0, load_bfrag(sWhh + ((t + 4) << 9)), hZ);
      hN = wmma16(ah0, load_bfrag(sWhh + ((t + 8) << 9)), hN);
      hR = wmma16(ah1, load_bfrag(sWhh + ((12 + t) << 9)), hR);
      hZ = wmma16(ah1, load_bfrag(sWhh + ((12 + t + 4) << 9)), hZ);
      hN = wmma16(ah1, load_bfrag(sWhh + ((12 + t + 8) << 9)), hN);
      int j = t * 16 + col;
#pragma unroll
      for (int e = 0; e < 8; ++e) {
        int r = rowLoc + e + khi * 8;
        int gr = rowBase + r;
        float rg = sigm(gR[e] + sBi[j] + hR[e] + sBh[j]);
        float zg = sigm(gZ[e] + sBi[j + 64] + hZ[e] + sBh[j + 64]);
        float ng = tanhf(gN[e] + sBi[j + 128] + rg * (hN[e] + sBh[j + 128]));
        float hp = hprev[(size_t)gr * HID + j];
        float h2 = (1.f - zg) * ng + zg * hp;
        hout[(size_t)gr * HID + j] = h2;
        dout_h[(size_t)gr * HID + j] = h2;     // second tuple output
        sHout[r * HID + j] = (_Float16)h2;
      }
    }
  }

  // ---- q / k / v projections: (16x64) @ (64x32) ----
  {
    v16h a0 = load_a_lds(sHout + rowLoc * HID, HID);
    v16h a1 = load_a_lds(sHout + rowLoc * HID + 32, HID);
    for (int nt = 0; nt < 2; ++nt) {
      v8f aq = {}, ak = {}, av = {};
      aq = wmma16(a0, load_bfrag(sWq + ((nt) << 9)), aq);
      ak = wmma16(a0, load_bfrag(sWk + ((nt) << 9)), ak);
      av = wmma16(a0, load_bfrag(sWv + ((nt) << 9)), av);
      aq = wmma16(a1, load_bfrag(sWq + ((2 + nt) << 9)), aq);
      ak = wmma16(a1, load_bfrag(sWk + ((2 + nt) << 9)), ak);
      av = wmma16(a1, load_bfrag(sWv + ((2 + nt) << 9)), av);
      int j = nt * 16 + col;
#pragma unroll
      for (int e = 0; e < 8; ++e) {
        int gr = rowBase + rowLoc + e + khi * 8;
        qo[(size_t)gr * ATT + j] = aq[e];
        ko[(size_t)gr * ATT + j] = ak[e];
        float vvl = av[e] + sBv[j];
        vo[(size_t)gr * ATT + j] = vvl > 0.f ? vvl : 0.f;
      }
    }
  }

  // ---- f16 copy of h_out for kernel 2 staging (b128 stores) ----
  __syncthreads();
  for (int i8 = tid * 8; i8 < 128 * HID; i8 += 256 * 8)
    *(v8h*)(hout16 + (size_t)rowBase * HID + i8) = *(const v8h*)(sHout + i8);
}

// ---------------------------------------------------------------------------
// Kernel 2: bidirectional 31-step pair-GRU with fused hard-attention logits.
// Key insight: for step m, the neighbor index idx(i,m) in {m, m+1} only ->
// per block (4 batches) only 8 distinct 64-wide rows are staged per step.
// dir 0 (fwd): store partial logit yf@W_hard[:64].  dir 1 (bwd): add
// yb@W_hard[64:], bias, gumbel, tau-softmax -> hard_w.
// ---------------------------------------------------------------------------
__global__ void k2_pair_bigru(
    const _Float16* __restrict__ hout16, const float* __restrict__ gumbel,
    const float* __restrict__ Wih_f, const float* __restrict__ Whh_f,
    const float* __restrict__ bih_f, const float* __restrict__ bhh_f,
    const float* __restrict__ Wih_b, const float* __restrict__ Whh_b,
    const float* __restrict__ bih_b, const float* __restrict__ bhh_b,
    const float* __restrict__ W_hard, const float* __restrict__ b_hard,
    float* __restrict__ part, float* __restrict__ hardw) {
  extern __shared__ char smem[];
  _Float16* sWih = (_Float16*)smem;            // 128*192 frag tiles (4x12)
  _Float16* sWhh = sWih + 128 * 192;           // 64*192  frag tiles (2x12)
  _Float16* sXf  = sWhh + HID * 192;           // 128*64 own-h half, row-major
  _Float16* sNb  = sXf + 128 * HID;            // 8*64 neighbor rows
  _Float16* sHa  = sNb + 8 * HID;              // 128*64
  _Float16* sHb  = sHa + 128 * HID;            // 128*64
  float* sBih = (float*)(sHb + 128 * HID);     // 192
  float* sBhh = sBih + 192;                    // 192
  float* sWhd = sBhh + 192;                    // 64*2

  const int tid = threadIdx.x;
  const int rowBase = blockIdx.x * 128;
  const int wave = tid >> 5, lane = tid & 31;
  const int rowLoc = wave * 16;
  const int khi = lane >> 4, col = lane & 15;
  // per-lane row identity (for neighbor selection in the A-fragment)
  const int laneRow = rowLoc + (lane & 15);
  const int ia = (rowBase + laneRow) & 31;           // agent id of this row
  const int nbBase = (laneRow >> 5) << 1;            // 2 slots per batch

  // static own-h half (pure f16 b128 copies)
  for (int i8 = tid * 8; i8 < 128 * HID; i8 += 256 * 8)
    *(v8h*)(sXf + i8) = *(const v8h*)(hout16 + (size_t)rowBase * HID + i8);

  for (int dir = 0; dir < 2; ++dir) {
    const float* bih = dir ? bih_b : bih_f;
    const float* bhh = dir ? bhh_b : bhh_f;
    __syncthreads();                            // previous dir done with LDS
    if (dir == 0) {
      stage_w_frag<128, 192>(Wih_f, sWih);
      stage_w_frag<HID, 192>(Whh_f, sWhh);
    } else {
      stage_w_frag<128, 192>(Wih_b, sWih);
      stage_w_frag<HID, 192>(Whh_b, sWhh);
    }
    for (int i = tid; i < 192; i += 256) { sBih[i] = bih[i]; sBhh[i] = bhh[i]; }
    for (int i = tid; i < 128; i += 256) sWhd[i] = W_hard[dir * 128 + i];
    for (int i8 = tid * 8; i8 < 128 * HID; i8 += 256 * 8)
      *(v8h*)(sHa + i8) = (v8h)(_Float16)0.f;   // h0 = 0
    __syncthreads();

    _Float16* hcur = sHa;
    _Float16* hnxt = sHb;
    for (int s = 0; s < NM; ++s) {
      int m = dir ? (NM - 1 - s) : s;
      // stage 8 neighbor rows: (batch bl, slot 0) = row m, (slot 1) = row m+1
      if (tid < 64) {
        int i8 = tid * 8;                        // 8*64 halves total
        int bl = i8 >> 7, slot = (i8 >> 6) & 1, c = i8 & 63;
        int gsrc = ((rowBase >> 5) + bl) * NAG + m + slot;
        *(v8h*)(sNb + i8) = *(const v8h*)(hout16 + (size_t)gsrc * HID + c);
      }
      __syncthreads();

      // hoisted A-fragments (invariant over the 4 output tiles)
      const _Float16* nbp = sNb + (nbBase + ((m < ia) ? 0 : 1)) * HID;
      v16h ax0 = load_a_row(sXf + laneRow * HID);
      v16h ax1 = load_a_row(sXf + laneRow * HID + 32);
      v16h ax2 = load_a_row(nbp);
      v16h ax3 = load_a_row(nbp + 32);
      v16h ah0 = load_a_row(hcur + laneRow * HID);
      v16h ah1 = load_a_row(hcur + laneRow * HID + 32);

      for (int t = 0; t < 4; ++t) {
        v8f gR = {}, gZ = {}, gN = {};
        gR = wmma16(ax0, load_bfrag(sWih + ((t) << 9)), gR);
        gZ = wmma16(ax0, load_bfrag(sWih + ((t + 4) << 9)), gZ);
        gN = wmma16(ax0, load_bfrag(sWih + ((t + 8) << 9)), gN);
        gR = wmma16(ax1, load_bfrag(sWih + ((12 + t) << 9)), gR);
        gZ = wmma16(ax1, load_bfrag(sWih + ((12 + t + 4) << 9)), gZ);
        gN = wmma16(ax1, load_bfrag(sWih + ((12 + t + 8) << 9)), gN);
        gR = wmma16(ax2, load_bfrag(sWih + ((24 + t) << 9)), gR);
        gZ = wmma16(ax2, load_bfrag(sWih + ((24 + t + 4) << 9)), gZ);
        gN = wmma16(ax2, load_bfrag(sWih + ((24 + t + 8) << 9)), gN);
        gR = wmma16(ax3, load_bfrag(sWih + ((36 + t) << 9)), gR);
        gZ = wmma16(ax3, load_bfrag(sWih + ((36 + t + 4) << 9)), gZ);
        gN = wmma16(ax3, load_bfrag(sWih + ((36 + t + 8) << 9)), gN);
        v8f hR = {}, hZ = {}, hN = {};
        hR = wmma16(ah0, load_bfrag(sWhh + ((t) << 9)), hR);
        hZ = wmma16(ah0, load_bfrag(sWhh + ((t + 4) << 9)), hZ);
        hN = wmma16(ah0, load_bfrag(sWhh + ((t + 8) << 9)), hN);
        hR = wmma16(ah1, load_bfrag(sWhh + ((12 + t) << 9)), hR);
        hZ = wmma16(ah1, load_bfrag(sWhh + ((12 + t + 4) << 9)), hZ);
        hN = wmma16(ah1, load_bfrag(sWhh + ((12 + t + 8) << 9)), hN);
        int j = t * 16 + col;
#pragma unroll
        for (int e = 0; e < 8; ++e) {
          int r = rowLoc + e + khi * 8;
          float rg = sigm(gR[e] + sBih[j] + hR[e] + sBhh[j]);
          float zg = sigm(gZ[e] + sBih[j + 64] + hZ[e] + sBhh[j + 64]);
          float ng = tanhf(gN[e] + sBih[j + 128] + rg * (hN[e] + sBhh[j + 128]));
          float hp = (float)hcur[r * HID + j];
          hnxt[r * HID + j] = (_Float16)((1.f - zg) * ng + zg * hp);
        }
      }
      __syncthreads();

      // fused hard-attention logits (per row), vectorized LDS reads
      if (tid < 128) {
        int r = tid, gr = rowBase + r;
        float l0 = 0.f, l1 = 0.f;
        const v8h* hp8 = (const v8h*)(hnxt + r * HID);
#pragma unroll
        for (int c8 = 0; c8 < 8; ++c8) {
          v8h hv8 = hp8[c8];
#pragma unroll
          for (int jj = 0; jj < 8; ++jj) {
            float hv = (float)hv8[jj];
            l0 += hv * sWhd[(c8 * 8 + jj) * 2];
            l1 += hv * sWhd[(c8 * 8 + jj) * 2 + 1];
          }
        }
        size_t pidx = ((size_t)gr * NM + m) * 2;
        if (dir == 0) {
          part[pidx] = l0;
          part[pidx + 1] = l1;
        } else {
          l0 += part[pidx] + b_hard[0] + gumbel[pidx];
          l1 += part[pidx + 1] + b_hard[1] + gumbel[pidx + 1];
          float mx = fmaxf(l0, l1);
          float e0 = __expf((l0 - mx) / TAUF);
          float e1 = __expf((l1 - mx) / TAUF);
          hardw[(size_t)gr * NM + m] = e1 / (e0 + e1);
        }
      }
      _Float16* tp = hcur; hcur = hnxt; hnxt = tp;
      __syncthreads();
    }
  }
}

// ---------------------------------------------------------------------------
// Kernel 3: soft attention gated by hard_w + decoder (tiny; VALU).
// ---------------------------------------------------------------------------
__global__ void k3_attn_decode(
    const float* __restrict__ hout, const float* __restrict__ q,
    const float* __restrict__ k, const float* __restrict__ v,
    const float* __restrict__ hardw, const float* __restrict__ W_dec,
    const float* __restrict__ b_dec, float* __restrict__ out) {
  __shared__ float sWd[(HID + ATT) * NA];
  __shared__ float sBd[NA];
  for (int i = threadIdx.x; i < (HID + ATT) * NA; i += blockDim.x)
    sWd[i] = W_dec[i];
  for (int i = threadIdx.x; i < NA; i += blockDim.x) sBd[i] = b_dec[i];
  __syncthreads();

  int row = blockIdx.x * blockDim.x + threadIdx.x;
  if (row >= ROWS) return;
  int b = row >> 5, ia = row & 31;

  float qv[ATT];
#pragma unroll
  for (int d = 0; d < ATT; ++d) qv[d] = q[(size_t)row * ATT + d];

  const float inv = 0.1767766953f;  // 1/sqrt(ATT)
  float sc[NM];
  float mx = -1e30f;
#pragma unroll
  for (int m = 0; m < NM; ++m) {
    int j = (m < ia) ? m : m + 1;
    const float* kr = k + (size_t)(b * NAG + j) * ATT;
    float s = 0.f;
#pragma unroll
    for (int d = 0; d < ATT; ++d) s += qv[d] * kr[d];
    s *= inv;
    sc[m] = s;
    mx = fmaxf(mx, s);
  }
  float den = 0.f;
#pragma unroll
  for (int m = 0; m < NM; ++m) { sc[m] = __expf(sc[m] - mx); den += sc[m]; }

  float xv[ATT];
#pragma unroll
  for (int d = 0; d < ATT; ++d) xv[d] = 0.f;
#pragma unroll
  for (int m = 0; m < NM; ++m) {
    int j = (m < ia) ? m : m + 1;
    float w = (sc[m] / den) * hardw[(size_t)row * NM + m];
    const float* vr = v + (size_t)(b * NAG + j) * ATT;
#pragma unroll
    for (int d = 0; d < ATT; ++d) xv[d] += vr[d] * w;
  }

  float hv[HID];
#pragma unroll
  for (int d = 0; d < HID; ++d) hv[d] = hout[(size_t)row * HID + d];
#pragma unroll
  for (int c = 0; c < NA; ++c) {
    float acc = sBd[c];
#pragma unroll
    for (int kk = 0; kk < HID; ++kk) acc += hv[kk] * sWd[kk * NA + c];
#pragma unroll
    for (int kk = 0; kk < ATT; ++kk) acc += xv[kk] * sWd[(HID + kk) * NA + c];
    out[(size_t)row * NA + c] = acc;
  }
}

// ---------------------------------------------------------------------------
extern "C" void kernel_launch(void* const* d_in, const int* in_sizes, int n_in,
                              void* d_out, int out_size, void* d_ws,
                              size_t ws_size, hipStream_t stream) {
  (void)in_sizes; (void)n_in; (void)out_size; (void)ws_size;
  const float* obs    = (const float*)d_in[0];
  const float* hidden = (const float*)d_in[1];
  const float* gumbel = (const float*)d_in[2];
  const float* W_enc  = (const float*)d_in[3];
  const float* b_enc  = (const float*)d_in[4];
  const float* Wih_c  = (const float*)d_in[5];
  const float* Whh_c  = (const float*)d_in[6];
  const float* bih_c  = (const float*)d_in[7];
  const float* bhh_c  = (const float*)d_in[8];
  const float* Wih_f  = (const float*)d_in[9];
  const float* Whh_f  = (const float*)d_in[10];
  const float* bih_f  = (const float*)d_in[11];
  const float* bhh_f  = (const float*)d_in[12];
  const float* Wih_b  = (const float*)d_in[13];
  const float* Whh_b  = (const float*)d_in[14];
  const float* bih_b  = (const float*)d_in[15];
  const float* bhh_b  = (const float*)d_in[16];
  const float* W_hard = (const float*)d_in[17];
  const float* b_hard = (const float*)d_in[18];
  const float* Wq     = (const float*)d_in[19];
  const float* Wk     = (const float*)d_in[20];
  const float* Wv     = (const float*)d_in[21];
  const float* b_v    = (const float*)d_in[22];
  const float* W_dec  = (const float*)d_in[23];
  const float* b_dec  = (const float*)d_in[24];

  float* out = (float*)d_out;          // [ROWS*NA] then [ROWS*HID]
  float* ws = (float*)d_ws;
  float* hout  = ws;                                   // ROWS*HID f32
  float* qbuf  = hout + (size_t)ROWS * HID;            // ROWS*ATT
  float* kbuf  = qbuf + (size_t)ROWS * ATT;
  float* vbuf  = kbuf + (size_t)ROWS * ATT;
  float* part  = vbuf + (size_t)ROWS * ATT;            // ROWS*NM*2
  float* hardw = part + (size_t)ROWS * NM * 2;         // ROWS*NM
  _Float16* hout16 = (_Float16*)(hardw + (size_t)ROWS * NM);  // ROWS*HID f16
  float* out_h = out + (size_t)ROWS * NA;

  const size_t lds1 =
      2 * (OBSD * HID + 2 * HID * 192 + 3 * HID * ATT + 3 * 128 * HID) +
      4 * (HID + 192 + 192 + ATT);
  const size_t lds2 =
      2 * (128 * 192 + HID * 192 + 128 * HID + 8 * HID + 2 * 128 * HID) +
      4 * (192 + 192 + 128);
  hipFuncSetAttribute((const void*)k1_encode_gru_qkv,
                      hipFuncAttributeMaxDynamicSharedMemorySize, (int)lds1);
  hipFuncSetAttribute((const void*)k2_pair_bigru,
                      hipFuncAttributeMaxDynamicSharedMemorySize, (int)lds2);

  k1_encode_gru_qkv<<<ROWS / 128, 256, lds1, stream>>>(
      obs, hidden, W_enc, b_enc, Wih_c, Whh_c, bih_c, bhh_c, Wq, Wk, Wv, b_v,
      hout, hout16, qbuf, kbuf, vbuf, out_h);
  k2_pair_bigru<<<ROWS / 128, 256, lds2, stream>>>(
      hout16, gumbel, Wih_f, Whh_f, bih_f, bhh_f, Wih_b, Whh_b, bih_b, bhh_b,
      W_hard, b_hard, part, hardw);
  k3_attn_decode<<<ROWS / 256, 256, 0, stream>>>(
      hout, qbuf, kbuf, vbuf, hardw, W_dec, b_dec, out);
}
